// SSMGaussianInterpolator_42408507081242
// MI455X (gfx1250) — matrix operand order repaired and checked by reading, not compiled, not run on hardware
//
#include <hip/hip_runtime.h>
#include <hip/hip_bf16.h>
#include <math.h>

typedef __attribute__((ext_vector_type(16))) _Float16 v16h;
typedef __attribute__((ext_vector_type(8)))  float    v8f;

#define GDIM 14
#define DM   128
#define SS   16
#define NL   2
#define NT   8      // timesteps
#define TM   16     // points per block
#define RB   (TM*NT) // 128 rows per block
#define NTH  128

// LDS strides (elements)
#define XNS 136     // xn f16 stride
#define YCS 264     // ycat f16 stride
#define WDS 136     // staged square / BC weight f16 stride
#define WOS 264     // staged out-weight f16 stride

// LDS layout (bytes)
#define OFF_H    0
#define OFF_XNH  (OFF_H   + RB*DM*4)
#define OFF_DT   (OFF_XNH + RB*XNS*2)
#define OFF_BC   (OFF_DT  + RB*DM*4)
#define OFF_YC   (OFF_BC  + RB*32*4)
#define OFF_WT   (OFF_YC  + RB*YCS*2)
#define OFF_MISC (OFF_WT  + DM*WOS*2)
#define SMEM_BYTES (OFF_MISC + 128*4*3 + TM*16*4)

// branch-free, numerically stable softplus on the fast transcendental pipes
__device__ __forceinline__ float softplus_fast(float v) {
  return fmaxf(v, 0.f) + __logf(1.f + __expf(-fabsf(v)));
}

// A fragment: 16x32 f16 tile, row-major in LDS (lane = M row; K contiguous: lanes 0-15 K=0..15, lanes 16-31 K=16..31)
__device__ __forceinline__ v16h frag_a(const _Float16* base, int ld, int lane) {
  int r  = lane & 15;
  int kb = (lane & 16) ? 16 : 0;
  const _Float16* row = base + r * ld;
  v16h f;
#pragma unroll
  for (int vg = 0; vg < 8; ++vg) {
    int kk = kb + 2*vg;
    f[2*vg]   = row[kk];
    f[2*vg+1] = row[kk+1];
  }
  return f;
}

// B fragment: 32x16 f16 tile, sourced from W^T stored row-major [N][K] in LDS.
// lanes 0-15: K=0..15 contiguous; lanes 16-31: K=16..31 (same k order as frag_a).
__device__ __forceinline__ v16h frag_b(const _Float16* base, int ld, int lane) {
  int n  = lane & 15;
  int kb = (lane & 16) ? 16 : 0;
  const _Float16* row = base + n * ld + kb;
  v16h f;
#pragma unroll
  for (int i = 0; i < 16; ++i) f[i] = row[i];
  return f;
}

#define WMMA_F16(acc, a, b) \
  (acc) = __builtin_amdgcn_wmma_f32_16x16x32_f16(false, (a), false, (b), (short)0, (acc), false, false)

__global__ void __launch_bounds__(NTH)
SSMGaussianInterpolator_kernel(
    const float* __restrict__ xyz, const float* __restrict__ scl,
    const float* __restrict__ rot, const float* __restrict__ opa,
    const float* __restrict__ col, const float* __restrict__ tsc,
    const float* __restrict__ tstamp,
    const float* __restrict__ in_W, const float* __restrict__ in_b,
    const float* __restrict__ ln_g, const float* __restrict__ ln_b,
    const float* __restrict__ dt_W, const float* __restrict__ dt_b,
    const float* __restrict__ B_W,  const float* __restrict__ C_W,
    const float* __restrict__ A_log, const float* __restrict__ D_skip,
    const float* __restrict__ out_W, const float* __restrict__ out_b,
    const float* __restrict__ te_W1, const float* __restrict__ te_b1,
    const float* __restrict__ te_W2, const float* __restrict__ te_b2,
    const float* __restrict__ op_W,  const float* __restrict__ op_b,
    float* __restrict__ out, int P)
{
  extern __shared__ char smem[];
  float*    hbuf  = (float*)(smem + OFF_H);     // [128][128]
  _Float16* xnh   = (_Float16*)(smem + OFF_XNH);// [128][XNS]
  float*    dtbuf = (float*)(smem + OFF_DT);    // [128][128]
  float*    bcbuf = (float*)(smem + OFF_BC);    // [128][32]  (B | C)
  _Float16* ycat  = (_Float16*)(smem + OFF_YC); // [128][YCS]
  char*     wtmem = smem + OFF_WT;              // weight staging union
  float*    rowm  = (float*)(smem + OFF_MISC);  // [128]
  float*    rowrs = rowm + 128;                 // [128]
  float*    tebuf = rowrs + 128;                // [128]
  float*    obuf  = tebuf + 128;                // [16][16]

  const int tid  = threadIdx.x;
  const int wv   = tid >> 5;
  const int lane = tid & 31;
  const int p0   = blockIdx.x * TM;

  // ------------- prologue: h = params @ in_W + in_b + posenc -------------
  {
    float* pbuf = (float*)wtmem;                 // [128][16]
    float* wIn  = (float*)(wtmem + RB*16*4);     // [15][128] row 14 = bias
    for (int idx = tid; idx < RB*GDIM; idx += NTH) {
      int r = idx / GDIM, f = idx % GDIM;
      int pg = p0 + (r >> 3), n = r & 7;
      int base = n * P + pg;
      float v;
      if      (f < 3)   v = xyz[base*3 + f];
      else if (f < 6)   v = scl[base*3 + (f-3)];
      else if (f < 10)  v = rot[base*4 + (f-6)];
      else if (f == 10) v = opa[base];
      else              v = col[base*3 + (f-11)];
      pbuf[r*16 + f] = v;
    }
    for (int idx = tid; idx < 15*DM; idx += NTH) {
      int f = idx >> 7, c = idx & 127;
      wIn[idx] = (f < GDIM) ? in_W[f*DM + c] : in_b[c];
    }
    __syncthreads();
    for (int idx = tid; idx < RB*DM; idx += NTH) {
      int r = idx >> 7, c = idx & 127, n = r & 7;
      float acc = wIn[GDIM*DM + c];
#pragma unroll
      for (int f = 0; f < GDIM; ++f) acc += pbuf[r*16 + f] * wIn[f*DM + c];
      float fr  = __expf(-9.210340371976184f * (float)(c & 63) * (1.0f/64.0f));
      float ang = tstamp[n] * fr;
      acc += (c < 64) ? __sinf(ang) : __cosf(ang);
      hbuf[idx] = acc;
    }
    __syncthreads();
  }

  // ------------- SSM layers -------------
  for (int l = 0; l < NL; ++l) {
    // layernorm stats (one row per thread)
    {
      const float* hr = hbuf + tid*DM;
      float s = 0.f, s2 = 0.f;
      for (int c = 0; c < DM; ++c) { float v = hr[c]; s += v; s2 += v*v; }
      float m = s * (1.0f/DM);
      float var = s2 * (1.0f/DM) - m*m;
      rowm[tid]  = m;
      rowrs[tid] = rsqrtf(fmaxf(var, 0.f) + 1e-5f);
    }
    __syncthreads();
    for (int idx = tid; idx < RB*DM; idx += NTH) {
      int r = idx >> 7, c = idx & 127;
      float xv = (hbuf[idx] - rowm[r]) * rowrs[r] * ln_g[l*DM + c] + ln_b[l*DM + c];
      xnh[r*XNS + c] = (_Float16)xv;
    }
    __syncthreads();

    for (int dir = 0; dir < 2; ++dir) {
      const int ldix = l*2 + dir;
      _Float16* wTd  = (_Float16*)wtmem;                    // dtW^T [128][WDS]
      _Float16* wTbc = (_Float16*)(wtmem + DM*WDS*2);       // [B^T;C^T] [32][WDS]
      {
        const float* W = dt_W + (size_t)ldix*DM*DM;
        for (int idx = tid; idx < DM*DM; idx += NTH) {
          int i = idx >> 7, j = idx & 127;
          wTd[j*WDS + i] = (_Float16)W[idx];
        }
        const float* Bw = B_W + (size_t)ldix*DM*SS;
        const float* Cw = C_W + (size_t)ldix*DM*SS;
        for (int idx = tid; idx < DM*SS; idx += NTH) {
          int i = idx >> 4, s = idx & 15;
          wTbc[s*WDS + i]      = (_Float16)Bw[idx];
          wTbc[(16+s)*WDS + i] = (_Float16)Cw[idx];
        }
      }
      __syncthreads();

      // dt = softplus(xn @ dtW + dt_b)  : 8x8 tiles of 16x16, K=128
      for (int t = wv; t < 64; t += 4) {
        int tr = t >> 3, tc = t & 7;
        v8f acc = {};
#pragma unroll
        for (int ks = 0; ks < DM; ks += 32) {
          v16h a = frag_a(xnh + tr*16*XNS + ks, XNS, lane);
          v16h b = frag_b(wTd + tc*16*WDS + ks, WDS, lane);
          WMMA_F16(acc, a, b);
        }
        int n = lane & 15, mh = (lane & 16) ? 8 : 0;
        float bias = dt_b[ldix*DM + tc*16 + n];
#pragma unroll
        for (int vg = 0; vg < 8; ++vg) {
          float v = acc[vg] + bias;
          dtbuf[(tr*16 + mh + vg)*DM + tc*16 + n] = softplus_fast(v);
        }
      }
      // [Bm | Cm] = xn @ [BW | CW] : 8x2 tiles, K=128
      for (int t = wv; t < 16; t += 4) {
        int tr = t >> 1, tc = t & 1;
        v8f acc = {};
#pragma unroll
        for (int ks = 0; ks < DM; ks += 32) {
          v16h a = frag_a(xnh + tr*16*XNS + ks, XNS, lane);
          v16h b = frag_b(wTbc + tc*16*WDS + ks, WDS, lane);
          WMMA_F16(acc, a, b);
        }
        int n = lane & 15, mh = (lane & 16) ? 8 : 0;
#pragma unroll
        for (int vg = 0; vg < 8; ++vg)
          bcbuf[(tr*16 + mh + vg)*32 + tc*16 + n] = acc[vg];
      }
      __syncthreads();

      // selective scan: thread owns (point, channel); state s[16] in registers
      for (int pair = tid; pair < TM*DM; pair += NTH) {
        int p = pair >> 7, d = pair & 127;
        const float* Al = A_log + (size_t)(ldix*DM + d)*SS;
        float Ac[SS];
#pragma unroll
        for (int j = 0; j < SS; ++j) Ac[j] = -__expf(Al[j]);
        float dski = D_skip[ldix*DM + d];
        float s[SS];
#pragma unroll
        for (int j = 0; j < SS; ++j) s[j] = 0.f;
        for (int stp = 0; stp < NT; ++stp) {
          int n   = dir ? (NT-1-stp) : stp;
          int row = p*NT + n;
          float dt  = dtbuf[row*DM + d];
          float x   = (float)xnh[row*XNS + d];
          float dtx = dt * x;
          float y   = dski * x;
#pragma unroll
          for (int j = 0; j < SS; ++j) {
            s[j] = __expf(Ac[j]*dt) * s[j] + dtx * bcbuf[row*32 + j];
            y   += s[j] * bcbuf[row*32 + 16 + j];
          }
          ycat[row*YCS + dir*DM + d] = (_Float16)y;
        }
      }
      __syncthreads();
    } // dir

    // h += ycat @ out_W + out_b : stage out_W^T [128][256] f16
    {
      _Float16* wTo = (_Float16*)wtmem;
      const float* W = out_W + (size_t)l*2*DM*DM; // [256][128]
      for (int idx = tid; idx < 2*DM*DM; idx += NTH) {
        int i = idx >> 7, j = idx & 127;
        wTo[j*WOS + i] = (_Float16)W[idx];
      }
      __syncthreads();
      for (int t = wv; t < 64; t += 4) {
        int tr = t >> 3, tc = t & 7;
        v8f acc = {};
#pragma unroll
        for (int ks = 0; ks < 2*DM; ks += 32) {
          v16h a = frag_a(ycat + tr*16*YCS + ks, YCS, lane);
          v16h b = frag_b(wTo + tc*16*WOS + ks, WOS, lane);
          WMMA_F16(acc, a, b);
        }
        int n = lane & 15, mh = (lane & 16) ? 8 : 0;
        float bias = out_b[l*DM + tc*16 + n];
#pragma unroll
        for (int vg = 0; vg < 8; ++vg)
          hbuf[(tr*16 + mh + vg)*DM + tc*16 + n] += acc[vg] + bias;
      }
      __syncthreads();
    }
  } // layers

  // ------------- epilogue: interp + te MLP + head + activations -------------
  float t  = tsc[0];
  float tn = t * (float)(NT - 1);
  int   il = (int)floorf(tn);
  il = il < 0 ? 0 : (il > NT-2 ? NT-2 : il);
  int   ih = il + 1;
  float alpha = tn - (float)il;

  { // te vector (same for all points): thread c computes one output channel
    float acc = te_b2[tid];
    for (int k = 0; k < DM; ++k) {
      float pre = t * te_W1[k] + te_b1[k];
      float sl  = pre / (1.f + __expf(-pre));
      acc += sl * te_W2[k*DM + tid];
    }
    tebuf[tid] = acc;
  }
  __syncthreads();

  for (int idx = tid; idx < TM*GDIM; idx += NTH) {
    int p = idx / GDIM, g = idx % GDIM;
    const float* hl = hbuf + (p*NT + il)*DM;
    const float* hh = hbuf + (p*NT + ih)*DM;
    float acc = op_b[g];
    for (int c = 0; c < DM; ++c) {
      float hi = (1.f - alpha)*hl[c] + alpha*hh[c] + tebuf[c];
      acc += hi * op_W[c*GDIM + g];
    }
    obuf[p*16 + g] = acc;
  }
  __syncthreads();

  if (tid < TM) {
    const float* o = obuf + tid*16;
    float r0 = o[6], r1 = o[7], r2 = o[8], r3 = o[9];
    float nrm = sqrtf(r0*r0 + r1*r1 + r2*r2 + r3*r3);
    float inv = 1.f / fmaxf(nrm, 1e-12f);
    float* dst = out + (size_t)(p0 + tid) * GDIM;
    dst[0] = o[0]; dst[1] = o[1]; dst[2] = o[2];
#pragma unroll
    for (int k = 3; k < 6; ++k) dst[k] = softplus_fast(o[k]);
    dst[6] = r0*inv; dst[7] = r1*inv; dst[8] = r2*inv; dst[9] = r3*inv;
#pragma unroll
    for (int k = 10; k < 14; ++k) { float v = o[k]; dst[k] = 1.f / (1.f + __expf(-v)); }
  }
}

extern "C" void kernel_launch(void* const* d_in, const int* in_sizes, int n_in,
                              void* d_out, int out_size, void* d_ws, size_t ws_size,
                              hipStream_t stream) {
  (void)n_in; (void)out_size; (void)d_ws; (void)ws_size;
  const float* xyz    = (const float*)d_in[0];
  const float* scl    = (const float*)d_in[1];
  const float* rot    = (const float*)d_in[2];
  const float* opa    = (const float*)d_in[3];
  const float* col    = (const float*)d_in[4];
  const float* tsc    = (const float*)d_in[5];
  const float* tstamp = (const float*)d_in[6];
  const float* in_W   = (const float*)d_in[7];
  const float* in_b   = (const float*)d_in[8];
  const float* ln_g   = (const float*)d_in[9];
  const float* ln_b   = (const float*)d_in[10];
  const float* dt_W   = (const float*)d_in[11];
  const float* dt_b   = (const float*)d_in[12];
  const float* B_W    = (const float*)d_in[13];
  const float* C_W    = (const float*)d_in[14];
  const float* A_log  = (const float*)d_in[15];
  const float* D_skip = (const float*)d_in[16];
  const float* out_W  = (const float*)d_in[17];
  const float* out_b  = (const float*)d_in[18];
  const float* te_W1  = (const float*)d_in[19];
  const float* te_b1  = (const float*)d_in[20];
  const float* te_W2  = (const float*)d_in[21];
  const float* te_b2  = (const float*)d_in[22];
  const float* op_W   = (const float*)d_in[23];
  const float* op_b   = (const float*)d_in[24];
  float* out = (float*)d_out;

  const int P = in_sizes[0] / (NT * 3);     // xyz is (N,1,P,3)
  const int nblk = (P + TM - 1) / TM;

  (void)hipFuncSetAttribute((const void*)SSMGaussianInterpolator_kernel,
                            hipFuncAttributeMaxDynamicSharedMemorySize, SMEM_BYTES);

  SSMGaussianInterpolator_kernel<<<nblk, NTH, SMEM_BYTES, stream>>>(
      xyz, scl, rot, opa, col, tsc, tstamp,
      in_W, in_b, ln_g, ln_b, dt_W, dt_b, B_W, C_W, A_log, D_skip,
      out_W, out_b, te_W1, te_b1, te_W2, te_b2, op_W, op_b,
      out, P);
}
